// MHA_1778116460686
// MI455X (gfx1250) — matrix operand order repaired and checked by reading, not compiled, not run on hardware
//
#include <hip/hip_runtime.h>

// ---------------------------------------------------------------------------
// CDNA5 (gfx1250, wave32) fused MHA, all matmuls on v_wmma_f32_16x16x32_f16.
// Stage 0: one-shot f32->f16 conversion; weights stored transposed [N][K].
// Stage 1: q = (Q@WQ)/8, k = K@WK -> [H,B,S,64] f16 ; v -> [H,B,64,S] f16.
// Stage 2: causal flash attention. Block = 4 waves = 64 query rows of one
//          (h,b). Each 32-key K/V tile is staged once into double-buffered
//          LDS and shared by all 4 waves (4x cut in L2 traffic). Online
//          softmax: row-max via single-instruction v_max_num_f32_dpp
//          row_xmask butterflies; row-sum via a 5th WMMA against ones.
// Stage 3: y = Ocat @ WO^T(f16), 16x64 strips per wave.
// Workspace: Qp/Kp/Vt/Oc (4x8MB) + X16 (8MB reused) + W16 (6MB) + WO16 (2MB) = 48MB.
// ---------------------------------------------------------------------------

typedef __attribute__((ext_vector_type(16))) _Float16 v16h;
typedef __attribute__((ext_vector_type(8)))  _Float16 v8h;
typedef __attribute__((ext_vector_type(8)))  float    v8f;

#define WMMA_F16(a, b, c) \
  __builtin_amdgcn_wmma_f32_16x16x32_f16(false, (a), false, (b), (short)0, (c), false, false)

static constexpr int HEADS  = 16;
static constexpr int DMODEL = 1024;
static constexpr int DHEAD  = 64;
static constexpr int SEQ    = 2048;
static constexpr int BATCH  = 2;
static constexpr int ROWS   = SEQ * BATCH;  // 4096

// A-fragment (16x32 f16, M=lane%16, g=lane/16), contiguous element map:
// e in [0,16): K = g*16+e. Two 16B loads.
__device__ __forceinline__ v16h load_a_frag16(const _Float16* row, int g) {
  const v8h lo = *(const v8h*)(row + g * 16);
  const v8h hi = *(const v8h*)(row + g * 16 + 8);
  v16h r;
#pragma unroll
  for (int e = 0; e < 8; ++e) { r[e] = lo[e]; r[8 + e] = hi[e]; }
  return r;
}

// One-instruction butterfly stage: m = max(m, lanes m ^ XM) within the 16-lane
// row, fused into a single VOP2-DPP16 (no mov, no canonicalize).
#define DPP_MAX(m, XM)                                                        \
  asm("v_max_num_f32_dpp %0, %0, %0 row_xmask:" #XM                           \
      " row_mask:0xf bank_mask:0xf bound_ctrl:1"                              \
      : "+v"(m));

__device__ __forceinline__ float row_max16(float m) {
  DPP_MAX(m, 1);
  DPP_MAX(m, 2);
  DPP_MAX(m, 4);
  DPP_MAX(m, 8);
  return m;
}

// ---------------------------------------------------------------------------
// Stage 0a: activation f32 -> f16 (layout preserved, [4096,1024]).
// ---------------------------------------------------------------------------
__global__ __launch_bounds__(256) void cvt_x_kernel(const float* __restrict__ X,
                                                    _Float16* __restrict__ X16) {
  const size_t i = ((size_t)blockIdx.x * 256 + threadIdx.x) * 8;
#pragma unroll
  for (int e = 0; e < 8; ++e) X16[i + e] = (_Float16)X[i + e];
}

// ---------------------------------------------------------------------------
// Stage 0b: weight f32 [m][K][N] -> f16 transposed [m][N][K] (K contiguous).
// ---------------------------------------------------------------------------
template <int KD, int ND>
__global__ __launch_bounds__(256) void cvt_w_kernel(const float* __restrict__ W,
                                                    _Float16* __restrict__ WT) {
  const size_t idx = (size_t)blockIdx.x * 256 + threadIdx.x;
  const int k = (int)(idx & (KD - 1));
  const size_t rest = idx >> 10;  // KD == 1024
  const int n = (int)(rest & (ND - 1));
  const size_t m = rest / ND;
  WT[idx] = (_Float16)W[(m * KD + k) * ND + n];
}

// ---------------------------------------------------------------------------
// Stage 1: projection GEMM. One wave -> 16x64 strip (full head width);
// one A-frag load feeds 4 WMMAs per K-chunk.
// ---------------------------------------------------------------------------
template <int TRANSPOSED>
__global__ __launch_bounds__(128) void proj_kernel(const _Float16* __restrict__ X16,
                                                   const _Float16* __restrict__ W16T,
                                                   _Float16* __restrict__ out,
                                                   float scale) {
  const int lane = threadIdx.x & 31;
  const int wv   = threadIdx.x >> 5;
  const int mt   = blockIdx.x * 4 + wv;  // 0..255 row tiles
  const int h    = blockIdx.y;
  const int g    = lane >> 4;
  const int nidx = lane & 15;

  const _Float16* __restrict__ arow = X16 + (size_t)(mt * 16 + nidx) * DMODEL;
  const _Float16* __restrict__ brow =
      W16T + ((size_t)h * DHEAD + nidx) * DMODEL + g * 16;

  const v8f vzero = {};
  v8f acc[4];
#pragma unroll
  for (int c = 0; c < 4; ++c) acc[c] = vzero;

  for (int kk = 0; kk < DMODEL / 32; ++kk) {
    const v16h af = load_a_frag16(arow + kk * 32, g);
#pragma unroll
    for (int c = 0; c < 4; ++c) {
      const v16h bf = *(const v16h*)(brow + (size_t)(c * 16) * DMODEL + kk * 32);
      acc[c] = WMMA_F16(af, bf, acc[c]);
    }
  }

#pragma unroll
  for (int c = 0; c < 4; ++c)
#pragma unroll
    for (int r = 0; r < 8; ++r) {
      const int R   = mt * 16 + r + 8 * g;  // = s*B + b
      const int s   = R >> 1;
      const int bb  = R & 1;
      const int col = c * 16 + nidx;
      const float v = acc[c][r] * scale;
      if (TRANSPOSED == 0)
        out[(((size_t)h * BATCH + bb) * SEQ + s) * DHEAD + col] = (_Float16)v;
      else
        out[(((size_t)h * BATCH + bb) * DHEAD + col) * SEQ + s] = (_Float16)v;
    }
}

// ---------------------------------------------------------------------------
// Stage 2: one flash-attention key tile (32 keys) read from LDS-staged tiles.
// MASKED only for the wave's diagonal tile.
// ---------------------------------------------------------------------------
template <bool MASKED>
__device__ __forceinline__ void attn_tile(int t, int qt, int g, int nidx,
                                          const _Float16 (*__restrict__ kls)[DHEAD],
                                          const _Float16 (*__restrict__ vls)[32],
                                          const v16h& qa0, const v16h& qa1,
                                          const v16h& ones,
                                          v8f (&acc)[4], v8f& lacc, v8f& mrow,
                                          _Float16 (*__restrict__ pls)[32]) {
  const v8f vzero = {};
  // B-fragments straight out of LDS (ds_load_b128 pairs, WMMA-operand shaped).
  const v16h kb00 = *(const v16h*)(&kls[nidx][g * 16]);
  const v16h kb01 = *(const v16h*)(&kls[nidx][32 + g * 16]);
  const v16h kb10 = *(const v16h*)(&kls[16 + nidx][g * 16]);
  const v16h kb11 = *(const v16h*)(&kls[16 + nidx][32 + g * 16]);
  v16h vb[4];
#pragma unroll
  for (int c = 0; c < 4; ++c)
    vb[c] = *(const v16h*)(&vls[c * 16 + nidx][g * 16]);

  v8f s0 = vzero, s1 = vzero;
  s0 = WMMA_F16(qa0, kb00, s0);
  s0 = WMMA_F16(qa1, kb01, s0);
  s1 = WMMA_F16(qa0, kb10, s1);
  s1 = WMMA_F16(qa1, kb11, s1);

  // Online softmax update (mask only on the diagonal tile).
  const int key0 = t * 32 + nidx;
  v8f p0, p1, sc;
#pragma unroll
  for (int r = 0; r < 8; ++r) {
    const int qi = qt * 16 + r + 8 * g;
    float a0 = s0[r];
    float a1 = s1[r];
    if (MASKED) {
      a0 = (key0 <= qi) ? a0 : -3.0e38f;
      a1 = (key0 + 16 <= qi) ? a1 : -3.0e38f;
    }
    const float rm = row_max16(fmaxf(a0, a1));
    const float mn = fmaxf(mrow[r], rm);
    sc[r] = __expf(mrow[r] - mn);
    p0[r] = __expf(a0 - mn);
    p1[r] = __expf(a1 - mn);
    mrow[r] = mn;
  }
  // Rescale O accumulators and the denominator accumulator.
#pragma unroll
  for (int c = 0; c < 4; ++c)
#pragma unroll
    for (int r = 0; r < 8; ++r) acc[c][r] *= sc[r];
#pragma unroll
  for (int r = 0; r < 8; ++r) lacc[r] *= sc[r];

  // P: C-layout -> A-layout via per-wave LDS tile (same-wave DS ordering).
#pragma unroll
  for (int r = 0; r < 8; ++r) {
    pls[r + 8 * g][nidx]      = (_Float16)p0[r];
    pls[r + 8 * g][16 + nidx] = (_Float16)p1[r];
  }
  asm volatile("s_wait_dscnt 0" ::: "memory");
  const v16h pa = load_a_frag16(&pls[nidx][0], g);

  // O += P @ V ; l += P @ 1 (row-sum computed by the matrix pipe).
#pragma unroll
  for (int c = 0; c < 4; ++c) acc[c] = WMMA_F16(pa, vb[c], acc[c]);
  lacc = WMMA_F16(pa, ones, lacc);
}

__global__ __launch_bounds__(128) void attn_kernel(const _Float16* __restrict__ Qp,
                                                   const _Float16* __restrict__ Kp,
                                                   const _Float16* __restrict__ Vt,
                                                   _Float16* __restrict__ Ocat) {
  __shared__ __align__(64) _Float16 Kls[2][32][DHEAD];  // 2 x 4KB
  __shared__ __align__(64) _Float16 Vls[2][DHEAD][32];  // 2 x 4KB
  __shared__ __align__(64) _Float16 Pls[4][16][32];     // 4 x 1KB
  const int tid  = threadIdx.x;
  const int lane = tid & 31;
  const int wv   = tid >> 5;
  const int qb   = blockIdx.x * 64;      // block's first query row
  const int qt   = blockIdx.x * 4 + wv;  // wave's 16-row query tile
  const int hb   = blockIdx.y;           // h*BATCH + b
  const int h    = hb >> 1;
  const int b    = hb & 1;
  const int g    = lane >> 4;
  const int nidx = lane & 15;

  const _Float16* __restrict__ qbase = Qp + (size_t)hb * SEQ * DHEAD;
  const _Float16* __restrict__ kbase = Kp + (size_t)hb * SEQ * DHEAD;
  const _Float16* __restrict__ vbase = Vt + (size_t)hb * DHEAD * SEQ;

  const _Float16* __restrict__ qrow = qbase + (size_t)(qt * 16 + nidx) * DHEAD;
  const v16h qa0 = load_a_frag16(qrow, g);
  const v16h qa1 = load_a_frag16(qrow + 32, g);

  v16h ones;
#pragma unroll
  for (int e = 0; e < 16; ++e) ones[e] = (_Float16)1.0f;

  const v8f vzero = {};
  v8f acc[4];
#pragma unroll
  for (int c = 0; c < 4; ++c) acc[c] = vzero;
  v8f lacc = vzero, mrow;
#pragma unroll
  for (int r = 0; r < 8; ++r) mrow[r] = -3.0e38f;

  // Staging split: 128 threads move one 32B line each of K (32x64) and V (64x32).
  const int krow = tid >> 2, kcol = (tid & 3) * 16;
  const int vrow = tid >> 1, vcol = (tid & 1) * 16;

  const int myDiag = (qt * 16 + 15) >> 5;   // wave's (masked) diagonal tile
  const int nktBlk = ((qb + 63) >> 5) + 1;  // block-uniform tile count

  for (int t = 0; t < nktBlk; ++t) {
    const int p = t & 1;
    // Stage this tile's K and V into LDS (shared by all 4 waves).
    *(v16h*)(&Kls[p][krow][kcol]) =
        *(const v16h*)(kbase + (size_t)(t * 32 + krow) * DHEAD + kcol);
    *(v16h*)(&Vls[p][vrow][vcol]) =
        *(const v16h*)(vbase + (size_t)vrow * SEQ + t * 32 + vcol);
    // Start the next tile's K lines moving toward the WGP during compute.
    if (t + 1 < nktBlk)
      __builtin_prefetch(kbase + (size_t)((t + 1) * 32 + krow) * DHEAD + kcol, 0, 3);
    __syncthreads();

    if (t < myDiag)
      attn_tile<false>(t, qt, g, nidx, Kls[p], Vls[p], qa0, qa1, ones,
                       acc, lacc, mrow, Pls[wv]);
    else if (t == myDiag)
      attn_tile<true>(t, qt, g, nidx, Kls[p], Vls[p], qa0, qa1, ones,
                      acc, lacc, mrow, Pls[wv]);
    // t > myDiag: barrier-only (at most one trailing tile per wave).
  }

  // Normalize; store heads-concatenated f16: Ocat[s, b, h*64 + dv].
#pragma unroll
  for (int r = 0; r < 8; ++r) {
    const float inv = 1.0f / lacc[r];
    const int s = qt * 16 + r + 8 * g;
    _Float16* __restrict__ orow =
        Ocat + ((size_t)s * BATCH + b) * (HEADS * DHEAD) + h * DHEAD;
#pragma unroll
    for (int c = 0; c < 4; ++c)
      orow[c * 16 + nidx] = (_Float16)(acc[c][r] * inv);
  }
}

// ---------------------------------------------------------------------------
// Stage 3: Y[4096,1024](f32) = Ocat(f16) @ WO16T(f16,[N][K]); 16x64 strips.
// ---------------------------------------------------------------------------
__global__ __launch_bounds__(128) void outproj_kernel(const _Float16* __restrict__ Ocat,
                                                      const _Float16* __restrict__ WO16T,
                                                      float* __restrict__ Y) {
  const int lane = threadIdx.x & 31;
  const int wv   = threadIdx.x >> 5;
  const int tid  = blockIdx.x * 4 + wv;  // 0..4095 strips
  const int mt   = tid >> 4;             // 0..255
  const int ns   = tid & 15;             // 64-col strip
  const int g    = lane >> 4;
  const int nidx = lane & 15;

  const _Float16* __restrict__ arow = Ocat + (size_t)(mt * 16 + nidx) * DMODEL;
  const _Float16* __restrict__ brow =
      WO16T + (size_t)(ns * 64 + nidx) * DMODEL + g * 16;

  const v8f vzero = {};
  v8f acc[4];
#pragma unroll
  for (int c = 0; c < 4; ++c) acc[c] = vzero;

  for (int kk = 0; kk < DMODEL / 32; ++kk) {
    const v16h af = load_a_frag16(arow + kk * 32, g);
#pragma unroll
    for (int c = 0; c < 4; ++c) {
      const v16h bf = *(const v16h*)(brow + (size_t)(c * 16) * DMODEL + kk * 32);
      acc[c] = WMMA_F16(af, bf, acc[c]);
    }
  }
#pragma unroll
  for (int c = 0; c < 4; ++c)
#pragma unroll
    for (int r = 0; r < 8; ++r) {
      const int R = mt * 16 + r + 8 * g;
      Y[(size_t)R * DMODEL + ns * 64 + c * 16 + nidx] = acc[c][r];
    }
}

// ---------------------------------------------------------------------------
extern "C" void kernel_launch(void* const* d_in, const int* in_sizes, int n_in,
                              void* d_out, int out_size, void* d_ws, size_t ws_size,
                              hipStream_t stream) {
  const float* Q  = (const float*)d_in[0];
  const float* K  = (const float*)d_in[1];
  const float* V  = (const float*)d_in[2];
  const float* WQ = (const float*)d_in[3];
  const float* WK = (const float*)d_in[4];
  const float* WV = (const float*)d_in[5];
  const float* WO = (const float*)d_in[6];
  float* Y = (float*)d_out;

  const size_t per = (size_t)HEADS * BATCH * SEQ * DHEAD;  // 4M f16
  const size_t wsz = (size_t)HEADS * DHEAD * DMODEL;       // 1M f16 per proj weight
  _Float16* Qp   = (_Float16*)d_ws;
  _Float16* Kp   = Qp + per;
  _Float16* Vt   = Kp + per;
  _Float16* Oc   = Vt + per;
  _Float16* X16  = Oc + per;          // 4M, reused for Q, K, V activations
  _Float16* WQ16 = X16 + per;         // [H][64][1024]
  _Float16* WK16 = WQ16 + wsz;
  _Float16* WV16 = WK16 + wsz;
  _Float16* WO16 = WV16 + wsz;        // [1024][1024] ([N][K])

  const dim3 blk128(128), blk256(256);
  const int xblocks = (int)(((size_t)ROWS * DMODEL) / 8 / 256);  // 2048
  const int wblocks = (int)(wsz / 256);                          // 4096
  const dim3 pgrid(64, HEADS);                                   // 256 strips/head

  // Weights (independent of activations).
  cvt_w_kernel<DMODEL, DHEAD><<<wblocks, blk256, 0, stream>>>(WQ, WQ16);
  cvt_w_kernel<DMODEL, DHEAD><<<wblocks, blk256, 0, stream>>>(WK, WK16);
  cvt_w_kernel<DMODEL, DHEAD><<<wblocks, blk256, 0, stream>>>(WV, WV16);
  cvt_w_kernel<DMODEL, DMODEL><<<(int)((size_t)DMODEL * DMODEL / 256), blk256, 0, stream>>>(WO, WO16);

  // Projections (X16 reuse is safe: same stream serializes).
  cvt_x_kernel<<<xblocks, blk256, 0, stream>>>(Q, X16);
  proj_kernel<0><<<pgrid, blk128, 0, stream>>>(X16, WQ16, Qp, 0.125f);  // 1/sqrt(64)
  cvt_x_kernel<<<xblocks, blk256, 0, stream>>>(K, X16);
  proj_kernel<0><<<pgrid, blk128, 0, stream>>>(X16, WK16, Kp, 1.0f);
  cvt_x_kernel<<<xblocks, blk256, 0, stream>>>(V, X16);
  proj_kernel<1><<<pgrid, blk128, 0, stream>>>(X16, WV16, Vt, 1.0f);

  attn_kernel<<<dim3(SEQ / 64, HEADS * BATCH), blk128, 0, stream>>>(Qp, Kp, Vt, Oc);

  outproj_kernel<<<dim3((ROWS / 16) * (DMODEL / 16 / 4) / 4), blk128, 0, stream>>>(Oc, WO16, Y);
}
